// LlamaAttention_46119358824781
// MI455X (gfx1250) — matrix operand order repaired and checked by reading, not compiled, not run on hardware
//
#include <hip/hip_runtime.h>

typedef _Float16 f16;
typedef __attribute__((ext_vector_type(16))) _Float16 v16h;
typedef __attribute__((ext_vector_type(8)))  _Float16 v8h;
typedef __attribute__((ext_vector_type(8)))  float    v8f;

#define HIDDEN 2048
#define SEQ    2048
#define NH     32
#define NKV    8
#define HD     64
#define LN_THETA 13.122363377404328f   /* ln(500000) */

// ---------------- fragment loaders (per CDNA5 ISA 7.12.2 layouts) ----------------

// A-matrix 16x32 f16: lane L holds row M=L&15; VGPR half = L>>4;
// element e: K = 16*(e>>3) + 8*half + (e&7)  -> two contiguous 8-f16 runs.
__device__ __forceinline__ v16h load_afrag(const f16* __restrict__ base, int ld,
                                           int row_base, int k_base, int lane) {
  int r  = row_base + (lane & 15);
  int hi = lane >> 4;
  const f16* p = base + (size_t)r * ld + k_base + 8 * hi;
  v8h lo = *(const v8h*)p;
  v8h h1 = *(const v8h*)(p + 16);
  v16h out;
#pragma unroll
  for (int e = 0; e < 8; e++) { out[e] = lo[e]; out[e + 8] = h1[e]; }
  return out;
}

// B-matrix 32x16 f16: lane L holds column N=L&15; half = L>>4;
// element e: K = 16*(e>>3) + 8*half + (e&7)  -> two contiguous 8-f16 runs.
// Source is stored N-major (row = output column, K contiguous).
__device__ __forceinline__ v16h load_bfrag(const f16* __restrict__ base, int ld,
                                           int n_base, int k_base, int lane) {
  int n  = n_base + (lane & 15);
  int hi = lane >> 4;
  const f16* p = base + (size_t)n * ld + k_base + 8 * hi;
  v8h lo = *(const v8h*)p;
  v8h h1 = *(const v8h*)(p + 16);
  v16h out;
#pragma unroll
  for (int e = 0; e < 8; e++) { out[e] = lo[e]; out[e + 8] = h1[e]; }
  return out;
}

#define WMMA_F16(A, B, C) \
  __builtin_amdgcn_wmma_f32_16x16x32_f16(false, (A), false, (B), (short)0, (C), false, false)

// ---------------- conversion kernels ----------------

__global__ __launch_bounds__(256) void cvt_kernel(const float* __restrict__ src,
                                                  f16* __restrict__ dst, int n) {
  int i = blockIdx.x * 256 + threadIdx.x;
  if (i < n) dst[i] = (f16)src[i];
}

// src: K x N (row-major f32). dst: N x K (row-major f16) == transposed.
__global__ __launch_bounds__(256) void cvt_t_kernel(const float* __restrict__ src,
                                                    f16* __restrict__ dst, int K, int N) {
  int i = blockIdx.x * 256 + threadIdx.x;
  if (i >= K * N) return;
  int n = i / K, k = i - n * K;
  dst[i] = (f16)src[(size_t)k * N + n];
}

// ---------------- projection GEMM: C[M,N] = A[M,K] @ Bt[N,K]^T ----------------
// Block tile 128x128 (8 waves as 4(M) x 2(N)); wave tile 32x64 (2x4 WMMA frags).
// mode 0: f32 row-major to outF
// mode 1: RoPE epilogue, f16 store to outRope[[col>>6]][pos][col&63]  (Q and K)
// mode 2: f16 store transposed: outVt[col][pos]                      (V)
__global__ __launch_bounds__(256) void gemm_kernel(const f16* __restrict__ A,
                                                   const f16* __restrict__ Bt,
                                                   int K, int N,
                                                   float* __restrict__ outF,
                                                   f16* __restrict__ outRope,
                                                   f16* __restrict__ outVt,
                                                   int mode) {
  int lane = threadIdx.x & 31;
  int wave = threadIdx.x >> 5;
  int wm = wave & 3, wn = wave >> 2;           // 4 waves along M, 2 along N
  int mo = blockIdx.y * 128 + wm * 32;
  int no = blockIdx.x * 128 + wn * 64;

  v8f acc[2][4];
#pragma unroll
  for (int i = 0; i < 2; i++)
#pragma unroll
    for (int j = 0; j < 4; j++) acc[i][j] = (v8f){};

  for (int kb = 0; kb < K; kb += 32) {
    v16h a0 = load_afrag(A, K, mo,      kb, lane);
    v16h a1 = load_afrag(A, K, mo + 16, kb, lane);
    v16h b[4];
#pragma unroll
    for (int j = 0; j < 4; j++) b[j] = load_bfrag(Bt, K, no + 16 * j, kb, lane);
#pragma unroll
    for (int j = 0; j < 4; j++) {
      acc[0][j] = WMMA_F16(a0, b[j], acc[0][j]);
      acc[1][j] = WMMA_F16(a1, b[j], acc[1][j]);
    }
  }

  int hi = lane >> 4, nl = lane & 15;
#pragma unroll
  for (int i = 0; i < 2; i++) {
#pragma unroll
    for (int j = 0; j < 4; j++) {
#pragma unroll
      for (int r = 0; r < 8; r++) {
        int gm = mo + 16 * i + r + 8 * hi;   // token position
        int gn = no + 16 * j + nl;           // output column
        float v = acc[i][j][r];
        if (mode == 0) {
          outF[(size_t)gm * N + gn] = v;
        } else if (mode == 1) {
          // RoPE: pair partner lives in the adjacent lane (column gn^1)
          float partner = __shfl_xor(v, 1, 32);
          int d = gn & (HD - 1);
          float freq = __expf(-(float)(d & ~1) * (LN_THETA / 64.0f));
          float ang = (float)gm * freq;
          float sn, cs;
          __sincosf(ang, &sn, &cs);
          float res = (d & 1) ? (partner * sn + v * cs) : (v * cs - partner * sn);
          outRope[((size_t)(gn >> 6) * SEQ + gm) * HD + d] = (f16)res;
        } else {
          outVt[(size_t)gn * SEQ + gm] = (f16)v;  // V transposed: [kv*64+d][pos]
        }
      }
    }
  }
}

// ---------------- flash attention ----------------
// Qh: [NH][SEQ][HD] f16 (rope applied), Kh: [NKV][SEQ][HD] f16 (rope applied),
// Vt: [NKV][HD][SEQ] f16.  Output: attn [SEQ][NH*HD] f16 row-major.
__global__ __launch_bounds__(256) void flash_kernel(const f16* __restrict__ Qh,
                                                    const f16* __restrict__ Kh,
                                                    const f16* __restrict__ Vt,
                                                    f16* __restrict__ attnOut) {
  __shared__ __align__(16) f16 lds[8][16 * 32];
  int lane = threadIdx.x & 31;
  int wave = threadIdx.x >> 5;
  int h = blockIdx.y;
  int kv = h >> 2;
  int qbase = (blockIdx.x * 8 + wave) * 16;
  int hi = lane >> 4, nl = lane & 15;

  const f16* Q  = Qh + (size_t)h * SEQ * HD;
  const f16* Kp = Kh + (size_t)kv * SEQ * HD;
  const f16* Vp = Vt + (size_t)kv * HD * SEQ;

  v16h qa0 = load_afrag(Q, HD, qbase, 0, lane);
  v16h qa1 = load_afrag(Q, HD, qbase, 32, lane);

  v8f o0 = {}, o1 = {}, o2 = {}, o3 = {};
  float mrow[8], lrow[8];
#pragma unroll
  for (int r = 0; r < 8; r++) { mrow[r] = -1e30f; lrow[r] = 0.0f; }

  const v8f zeroC = {};
  for (int kb = 0; kb <= qbase + 15; kb += 32) {
    // S = Q @ K^T for 32 keys (two 16-wide column tiles, K-dim = 64 in 2 steps)
    v16h k00 = load_bfrag(Kp, HD, kb,      0,  lane);
    v16h k01 = load_bfrag(Kp, HD, kb,      32, lane);
    v16h k10 = load_bfrag(Kp, HD, kb + 16, 0,  lane);
    v16h k11 = load_bfrag(Kp, HD, kb + 16, 32, lane);
    v8f s0 = WMMA_F16(qa0, k00, zeroC);
    s0     = WMMA_F16(qa1, k01, s0);
    v8f s1 = WMMA_F16(qa0, k10, zeroC);
    s1     = WMMA_F16(qa1, k11, s1);

    float fac[8];
#pragma unroll
    for (int r = 0; r < 8; r++) {
      int qrow = qbase + r + 8 * hi;
      float e0 = s0[r] * 0.125f;                     // 1/sqrt(64)
      float e1 = s1[r] * 0.125f;
      if (kb + nl      > qrow) e0 = -1e30f;          // causal mask
      if (kb + 16 + nl > qrow) e1 = -1e30f;
      float rmax = fmaxf(e0, e1);                    // row-max across 16 lanes of half
      rmax = fmaxf(rmax, __shfl_xor(rmax, 1, 32));
      rmax = fmaxf(rmax, __shfl_xor(rmax, 2, 32));
      rmax = fmaxf(rmax, __shfl_xor(rmax, 4, 32));
      rmax = fmaxf(rmax, __shfl_xor(rmax, 8, 32));
      float mnew = fmaxf(mrow[r], rmax);
      float f = __expf(mrow[r] - mnew);
      float p0 = __expf(e0 - mnew);
      float p1 = __expf(e1 - mnew);
      float ps = p0 + p1;
      ps += __shfl_xor(ps, 1, 32);
      ps += __shfl_xor(ps, 2, 32);
      ps += __shfl_xor(ps, 4, 32);
      ps += __shfl_xor(ps, 8, 32);
      lrow[r] = lrow[r] * f + ps;
      mrow[r] = mnew;
      fac[r] = f;
      int m = r + 8 * hi;                            // stage P into LDS (C -> A reshape)
      lds[wave][m * 32 + nl]      = (f16)p0;
      lds[wave][m * 32 + 16 + nl] = (f16)p1;
    }
#pragma unroll
    for (int r = 0; r < 8; r++) {
      o0[r] *= fac[r]; o1[r] *= fac[r]; o2[r] *= fac[r]; o3[r] *= fac[r];
    }

    asm volatile("s_wait_dscnt 0" ::: "memory");     // per-wave LDS RAW fence

    // reload P as an A fragment (16x32, K-dim = keys)
    v16h pa;
    {
      const f16* lp = &lds[wave][nl * 32 + 8 * hi];
      v8h lo = *(const v8h*)lp;
      v8h h1 = *(const v8h*)(lp + 16);
#pragma unroll
      for (int e = 0; e < 8; e++) { pa[e] = lo[e]; pa[e + 8] = h1[e]; }
    }

    // O += P @ V  (V transposed storage -> contiguous B fragments)
    v16h vb0 = load_bfrag(Vp, SEQ, 0,  kb, lane);
    v16h vb1 = load_bfrag(Vp, SEQ, 16, kb, lane);
    v16h vb2 = load_bfrag(Vp, SEQ, 32, kb, lane);
    v16h vb3 = load_bfrag(Vp, SEQ, 48, kb, lane);
    o0 = WMMA_F16(pa, vb0, o0);
    o1 = WMMA_F16(pa, vb1, o1);
    o2 = WMMA_F16(pa, vb2, o2);
    o3 = WMMA_F16(pa, vb3, o3);
  }

  // normalize and store: attn[pos][h*64 + d]
#pragma unroll
  for (int r = 0; r < 8; r++) {
    int gm = qbase + r + 8 * hi;
    float inv = 1.0f / lrow[r];
    size_t rowoff = (size_t)gm * (NH * HD) + (size_t)h * HD + nl;
    attnOut[rowoff +  0] = (f16)(o0[r] * inv);
    attnOut[rowoff + 16] = (f16)(o1[r] * inv);
    attnOut[rowoff + 32] = (f16)(o2[r] * inv);
    attnOut[rowoff + 48] = (f16)(o3[r] * inv);
  }
}

// ---------------- host side ----------------

extern "C" void kernel_launch(void* const* d_in, const int* in_sizes, int n_in,
                              void* d_out, int out_size, void* d_ws, size_t ws_size,
                              hipStream_t stream) {
  const float* hs = (const float*)d_in[0];
  // d_in[1] = attention_mask (causal; recomputed on the fly, unused)
  const float* wq = (const float*)d_in[2];
  const float* wk = (const float*)d_in[3];
  const float* wv = (const float*)d_in[4];
  const float* wo = (const float*)d_in[5];
  float* out = (float*)d_out;

  f16* base = (f16*)d_ws;
  size_t off = 0;
  f16* hsF  = base + off; off += (size_t)SEQ * HIDDEN;          // hidden f16
  f16* wqT  = base + off; off += (size_t)HIDDEN * (NH * HD);    // wq^T  [N][K]
  f16* wkT  = base + off; off += (size_t)HIDDEN * (NKV * HD);   // wk^T
  f16* wvT  = base + off; off += (size_t)HIDDEN * (NKV * HD);   // wv^T
  f16* woT  = base + off; off += (size_t)HIDDEN * HIDDEN;       // wo^T
  f16* Qh   = base + off; off += (size_t)NH * SEQ * HD;         // [h][pos][d]
  f16* Kh   = base + off; off += (size_t)NKV * SEQ * HD;        // [kv][pos][d]
  f16* VtA  = base + off; off += (size_t)NKV * HD * SEQ;        // [kv][d][pos]
  f16* attn = base + off; off += (size_t)SEQ * NH * HD;         // [pos][h*64+d]
  (void)ws_size; (void)in_sizes; (void)n_in; (void)out_size;

  const int T = 256;
  // 1) convert / transpose to f16
  cvt_kernel<<<(SEQ * HIDDEN + T - 1) / T, T, 0, stream>>>(hs, hsF, SEQ * HIDDEN);
  cvt_t_kernel<<<(HIDDEN * NH * HD  + T - 1) / T, T, 0, stream>>>(wq, wqT, HIDDEN, NH * HD);
  cvt_t_kernel<<<(HIDDEN * NKV * HD + T - 1) / T, T, 0, stream>>>(wk, wkT, HIDDEN, NKV * HD);
  cvt_t_kernel<<<(HIDDEN * NKV * HD + T - 1) / T, T, 0, stream>>>(wv, wvT, HIDDEN, NKV * HD);
  cvt_t_kernel<<<(HIDDEN * HIDDEN   + T - 1) / T, T, 0, stream>>>(wo, woT, HIDDEN, HIDDEN);

  // 2) projections (WMMA), RoPE fused into Q/K epilogue, V stored transposed
  gemm_kernel<<<dim3((NH * HD) / 128,  SEQ / 128), T, 0, stream>>>(
      hsF, wqT, HIDDEN, NH * HD, nullptr, Qh, nullptr, 1);
  gemm_kernel<<<dim3((NKV * HD) / 128, SEQ / 128), T, 0, stream>>>(
      hsF, wkT, HIDDEN, NKV * HD, nullptr, Kh, nullptr, 1);
  gemm_kernel<<<dim3((NKV * HD) / 128, SEQ / 128), T, 0, stream>>>(
      hsF, wvT, HIDDEN, NKV * HD, nullptr, nullptr, VtA, 2);

  // 3) causal flash attention (WMMA QK^T and PV, online softmax)
  flash_kernel<<<dim3(SEQ / 128, NH), T, 0, stream>>>(Qh, Kh, VtA, attn);

  // 4) output projection -> f32 out
  gemm_kernel<<<dim3(HIDDEN / 128, SEQ / 128), T, 0, stream>>>(
      attn, woT, HIDDEN, HIDDEN, out, nullptr, nullptr, 0);
}
